// NA_489626272367
// MI455X (gfx1250) — matrix operand order repaired and checked by reading, not compiled, not run on hardware
//
#include <hip/hip_runtime.h>

typedef _Float16 f16;
typedef __attribute__((ext_vector_type(16))) _Float16 v16h;
typedef __attribute__((ext_vector_type(8)))  _Float16 v8h;
typedef __attribute__((ext_vector_type(4)))  _Float16 v4h;
typedef __attribute__((ext_vector_type(8)))  float    v8f;
typedef __attribute__((ext_vector_type(4)))  float    v4f;
typedef __attribute__((ext_vector_type(4)))  unsigned int u32x4;
typedef __attribute__((ext_vector_type(8)))  int i32x8;
typedef __attribute__((ext_vector_type(4)))  int i32x4;

#define B_   4
#define C_   256
#define L_   4096
#define LH   2048          // per-parity class length
#define KS   65
#define RAD  32
#define SCALE_ 0.0625f     // 1/sqrt(256)
#define XP   264           // padded X row (f16), 528B = 33*16
#define VTP  152           // padded V^T row (f16), 304B = 19*16

// ---------------- WMMA helpers (layouts per CDNA5 ISA) ----------------
__device__ __forceinline__ v8f wmma16(v16h a, v16h b, v8f c) {
    return __builtin_amdgcn_wmma_f32_16x16x32_f16(false, a, false, b, (short)0, c,
                                                  false, false);
}
// A 16x32 f16: lane m=lane%16,h=lane/16; elems 0..15 = row[k0+16h .. k0+16h+15] (contiguous)
__device__ __forceinline__ v16h fragA(const f16* row, int k0, int h) {
    v8h lo = *(const v8h*)(row + k0 + 16*h);
    v8h hi = *(const v8h*)(row + k0 + 16*h + 8);
    v16h r;
#pragma unroll
    for (int i = 0; i < 8; ++i) { r[i] = lo[i]; r[i + 8] = hi[i]; }
    return r;
}
// B 32x16 f16: lane n=lane%16,h=lane/16; elems 0..15 = col[k0+16h .. k0+16h+15] (contiguous)
__device__ __forceinline__ v16h fragB(const f16* col, int k0, int h) {
    v8h lo = *(const v8h*)(col + k0 + 16*h);
    v8h hi = *(const v8h*)(col + k0 + 16*h + 8);
    v16h r;
#pragma unroll
    for (int i = 0; i < 8; ++i) { r[i] = lo[i]; r[i + 8] = hi[i]; }
    return r;
}
__device__ __forceinline__ v16h fragB16(const f16* p) {   // 16 contiguous f16
    v8h lo = *(const v8h*)(p);
    v8h hi = *(const v8h*)(p + 8);
    v16h r;
#pragma unroll
    for (int i = 0; i < 8; ++i) { r[i] = lo[i]; r[i + 8] = hi[i]; }
    return r;
}

// ---------------- Tensor Data Mover: 2-D row-block copy global->LDS ----------------
// 6-arg builtin on this toolchain:
//   (uint32x4 g0, int32x8 g1, int32x4 g2, int32x4 g3, int32x8 extra, i32 cpol)
// D# fields per CDNA5 ISA. Groups 2/3 zero => 2-D tensor.
#if __has_builtin(__builtin_amdgcn_tensor_load_to_lds)
#define HAS_TDM 1
__device__ __forceinline__ void tdm_load_rows(void* lds_dst, const void* gsrc,
                                              unsigned rows, unsigned rowlen /*f16 elems*/) {
    unsigned long long ga = (unsigned long long)gsrc;
    u32x4 g0;
    g0[0] = 1u;                                             // count=1, user descriptor
    g0[1] = (unsigned)(unsigned long long)lds_dst;          // lds_addr = low 32 bits
    g0[2] = (unsigned)ga;                                   // global_addr[31:0]
    g0[3] = (unsigned)((ga >> 32) & 0x01FFFFFFull) | (2u << 30); // addr[56:32] | type=2
    i32x8 g1;
    g1[0] = 1 << 16;                                        // wg_mask=0, data_size=1 (2B)
    g1[1] = (int)(rowlen << 16);                            // tensor_dim0[15:0] @63:48
    g1[2] = (int)((rowlen >> 16) | (rows << 16));           // dim0 hi | tensor_dim1 lo
    g1[3] = (int)((rows >> 16) | (rowlen << 16));           // dim1 hi | tile_dim0
    g1[4] = (int)rows;                                      // tile_dim1 (tile_dim2=0)
    g1[5] = (int)rowlen;                                    // tensor_dim0_stride lo32
    g1[6] = 0;                                              // stride hi | dim1_stride lo
    g1[7] = 0;
    i32x4 gz  = {0, 0, 0, 0};
    i32x8 gz8 = {0, 0, 0, 0, 0, 0, 0, 0};
    __builtin_amdgcn_tensor_load_to_lds(g0, g1, gz, gz, gz8, 0);
}
#else
#define HAS_TDM 0
#endif

__device__ __forceinline__ int cstart(int i) {              // dilation-1 class window start
    return i < RAD ? 0 : (i > (LH - KS) ? (LH - KS) : i - RAD);
}

// ============================ Stage 1: fused QKV projection ============================
// grid (L/64, B, 3), block 256.  D[d,l] = sum_c W[d,c]*X[c,l] + bias[d], stored f16 as
// ws[(b,parity,i,c)] with l = 2i+parity.
__global__ __launch_bounds__(256)
void natt_qkv(const float* __restrict__ x,
              const float* __restrict__ wq, const float* __restrict__ bq,
              const float* __restrict__ wk, const float* __restrict__ bk,
              const float* __restrict__ wv, const float* __restrict__ bv,
              f16* __restrict__ qo, f16* __restrict__ ko, f16* __restrict__ vo) {
    extern __shared__ char smem[];
    f16* Wl = (f16*)smem;                 // 256x256 f16 (A operand, row d contiguous c)
    f16* Xl = Wl + C_ * C_;               // 64 x XP f16 (B operand, row l contiguous c)

    const int l0 = blockIdx.x * 64;
    const int b  = blockIdx.y;
    const int z  = blockIdx.z;
    const float* w  = (z == 0) ? wq : (z == 1) ? wk : wv;
    const float* bs = (z == 0) ? bq : (z == 1) ? bk : bv;
    f16*         dst = (z == 0) ? qo : (z == 1) ? ko : vo;
    const int tid = threadIdx.x;

    // W -> LDS (f32 -> f16)
    for (int e = tid * 4; e < C_ * C_; e += 256 * 4) {
        float4 f = *(const float4*)(w + e);
        v4h h4 = { (f16)f.x, (f16)f.y, (f16)f.z, (f16)f.w };
        *(v4h*)(Wl + e) = h4;
    }
    // X tile -> LDS transposed [l][c] (f32 -> f16)
    for (int e = tid * 4; e < C_ * 64; e += 256 * 4) {
        const int c = e >> 6, lo = e & 63;
        float4 f = *(const float4*)(x + ((size_t)b * C_ + c) * L_ + l0 + lo);
        Xl[(lo + 0) * XP + c] = (f16)f.x;
        Xl[(lo + 1) * XP + c] = (f16)f.y;
        Xl[(lo + 2) * XP + c] = (f16)f.z;
        Xl[(lo + 3) * XP + c] = (f16)f.w;
    }
    __syncthreads();

    const int wid = tid >> 5, lane = tid & 31;
    const int n = lane & 15, h = lane >> 4;
#pragma unroll
    for (int dt = 0; dt < 2; ++dt) {
        const int dbase = (wid * 2 + dt) * 16;
        const f16* arow = Wl + (dbase + n) * C_;
        // hoist the 8 weight fragments for this d-tile (reused by 4 l-subtiles)
        v16h wa[8];
#pragma unroll
        for (int kk = 0; kk < 8; ++kk) wa[kk] = fragA(arow, kk * 32, h);
#pragma unroll
        for (int ls = 0; ls < 4; ++ls) {
            const f16* brow = Xl + (ls * 16 + n) * XP;
            // batch all B fragments first -> single dscnt wait, then 8 chained WMMAs
            v16h xb[8];
#pragma unroll
            for (int kk = 0; kk < 8; ++kk) xb[kk] = fragB(brow, kk * 32, h);
            v8f acc = {0.f, 0.f, 0.f, 0.f, 0.f, 0.f, 0.f, 0.f};
#pragma unroll
            for (int kk = 0; kk < 8; ++kk) acc = wmma16(wa[kk], xb[kk], acc);
            // D element (M=d=dbase+r+8h, N=l=l0+ls*16+n); add bias, pack to f16
            v8h oh;
#pragma unroll
            for (int r = 0; r < 8; ++r)
                oh[r] = (f16)(acc[r] + bs[dbase + 8 * h + r]);
            const int l = l0 + ls * 16 + n;
            f16* op = dst + ((size_t)((b * 2 + (l & 1)) * LH + (l >> 1))) * C_
                          + dbase + 8 * h;
            *(v8h*)op = oh;    // 16B contiguous store (d = dbase+8h .. +7)
        }
    }
}

// ============================ Stage 2: neighborhood attention ============================
// grid (LH/64, 2, B), block 128 (4 waves). WG handles 64 class queries of one (b,parity).
__global__ __launch_bounds__(128)
void natt_attn(const f16* __restrict__ qw, const f16* __restrict__ kw,
               const f16* __restrict__ vw, float* __restrict__ out) {
    extern __shared__ char smem[];
    f16* Kl = (f16*)smem;                 // 128 x 256
    f16* Vl = Kl + 128 * C_;              // 128 x 256 (reused later as O staging, f32)
    f16* Vt = Vl + 128 * C_;              // 256 x VTP (V transposed, zero-padded cols)
    f16* Ql = Vt + C_ * VTP;              // 64 x 256
    f16* Sl = Ql + 64 * C_;               // 4 x 16 x 96 (softmax probs, f16)
    float* Ol = (float*)Vl;               // 256 x 64 f32

    const int i0 = blockIdx.x * 64;
    const int p  = blockIdx.y;
    const int b  = blockIdx.z;
    const int tid = threadIdx.x, wid = tid >> 5, lane = tid & 31;
    const int n = lane & 15, h = lane >> 4;
    const size_t rb = (size_t)(b * 2 + p) * LH;
    int s0 = i0 - RAD;
    s0 = s0 < 0 ? 0 : s0;
    s0 = s0 > LH - 128 ? LH - 128 : s0;    // 128-row K/V window covers all 64 queries

#if HAS_TDM
    if (wid == 0) {
        tdm_load_rows(Kl, kw + (rb + s0) * C_, 128, C_);
        tdm_load_rows(Vl, vw + (rb + s0) * C_, 128, C_);
        tdm_load_rows(Ql, qw + (rb + i0) * C_, 64, C_);
        __builtin_amdgcn_s_wait_tensorcnt(0);
    }
#else
    for (int e = tid * 8; e < 128 * C_; e += 128 * 8) {
        *(v8h*)(Kl + e) = *(const v8h*)(kw + (rb + s0) * C_ + e);
        *(v8h*)(Vl + e) = *(const v8h*)(vw + (rb + s0) * C_ + e);
    }
    for (int e = tid * 8; e < 64 * C_; e += 128 * 8)
        *(v8h*)(Ql + e) = *(const v8h*)(qw + (rb + i0) * C_ + e);
#endif
    __syncthreads();

    // V -> V^T in LDS; zero-fill pad columns so zero-weight WMMA reads see finite zeros
    for (int e = tid * 8; e < 128 * C_; e += 128 * 8) {
        const int pos = e >> 8, c = e & 255;
        v8h vv = *(const v8h*)(Vl + e);
#pragma unroll
        for (int j = 0; j < 8; ++j) Vt[(c + j) * VTP + pos] = vv[j];
    }
    for (int e = tid; e < C_ * (VTP - 128); e += 128)
        Vt[(e / (VTP - 128)) * VTP + 128 + (e % (VTP - 128))] = (f16)0.f;
    __syncthreads();

    // ---- scores: S[q,kr] = sum_c Q[q,c] K[kr,c], 16 queries x 96 keys per wave ----
    const int q0 = i0 + 16 * wid;
    const int nb = (cstart(q0) - s0) & ~7;        // 8-aligned window base (16B LDS align)

    const f16* qrow = Ql + (16 * wid + n) * C_;   // A operand: m = lane%16
    v16h qa[8];
#pragma unroll
    for (int kk = 0; kk < 8; ++kk) qa[kk] = fragA(qrow, kk * 32, h);

    v8f sc[6];
#pragma unroll
    for (int nt = 0; nt < 6; ++nt) {
        int krow = nb + nt * 16 + n;
        krow = krow > 127 ? 127 : krow;           // clamped rows are masked below
        const f16* krp = Kl + krow * C_;
        // batch all B fragments first -> single dscnt wait, then 8 chained WMMAs
        v16h kb[8];
#pragma unroll
        for (int kk = 0; kk < 8; ++kk) kb[kk] = fragB(krp, kk * 32, h);
        v8f acc = {0.f, 0.f, 0.f, 0.f, 0.f, 0.f, 0.f, 0.f};
#pragma unroll
        for (int kk = 0; kk < 8; ++kk) acc = wmma16(qa[kk], kb[kk], acc);
        sc[nt] = acc;
    }
    // mask to per-query 65-key window, apply scale
#pragma unroll
    for (int nt = 0; nt < 6; ++nt)
#pragma unroll
        for (int r = 0; r < 8; ++r) {
            const int iq  = q0 + r + 8 * h;       // D row M = r + 8h
            const int off = cstart(iq) - s0;
            const int kr  = nb + nt * 16 + n;
            const bool ok = (kr >= off) && (kr <= off + 64);
            sc[nt][r] = ok ? sc[nt][r] * SCALE_ : -1e30f;
        }
    // softmax in registers: 16-lane xor-shuffle reductions (row q = (r,h), keys = lanes)
    v8f mx = sc[0];
#pragma unroll
    for (int nt = 1; nt < 6; ++nt)
#pragma unroll
        for (int r = 0; r < 8; ++r) mx[r] = fmaxf(mx[r], sc[nt][r]);
#pragma unroll
    for (int d = 1; d < 16; d <<= 1)
#pragma unroll
        for (int r = 0; r < 8; ++r) mx[r] = fmaxf(mx[r], __shfl_xor(mx[r], d, 32));
    v8f sm = {0.f, 0.f, 0.f, 0.f, 0.f, 0.f, 0.f, 0.f};
    v8f pw[6];
#pragma unroll
    for (int nt = 0; nt < 6; ++nt)
#pragma unroll
        for (int r = 0; r < 8; ++r) {
            const float e = __expf(sc[nt][r] - mx[r]);   // masked -> exp(-huge) = 0
            pw[nt][r] = e;
            sm[r] += e;
        }
#pragma unroll
    for (int d = 1; d < 16; d <<= 1)
#pragma unroll
        for (int r = 0; r < 8; ++r) sm[r] += __shfl_xor(sm[r], d, 32);
    float inv[8];
#pragma unroll
    for (int r = 0; r < 8; ++r) inv[r] = 1.f / sm[r];

    f16* srow0 = Sl + wid * 16 * 96;
#pragma unroll
    for (int nt = 0; nt < 6; ++nt)
#pragma unroll
        for (int r = 0; r < 8; ++r)
            srow0[(r + 8 * h) * 96 + nt * 16 + n] = (f16)(pw[nt][r] * inv[r]);

    // ---- O[q,c] = P (16x96) x V (96x256) ----
    const f16* prow = srow0 + n * 96;             // A operand: m = wave-local q
    v16h pa[3];
#pragma unroll
    for (int kk = 0; kk < 3; ++kk) pa[kk] = fragA(prow, kk * 32, h);
#pragma unroll
    for (int ct = 0; ct < 16; ++ct) {
        const f16* vtr = Vt + (ct * 16 + n) * VTP;   // B operand: n = c (lane%16)
        // batch the 3 B fragments, then 3 chained WMMAs
        v16h vb[3];
#pragma unroll
        for (int kk = 0; kk < 3; ++kk)
            vb[kk] = fragB16(vtr + nb + kk * 32 + 16 * h);
        v8f acc = {0.f, 0.f, 0.f, 0.f, 0.f, 0.f, 0.f, 0.f};
#pragma unroll
        for (int kk = 0; kk < 3; ++kk) acc = wmma16(pa[kk], vb[kk], acc);
        float* od = Ol + (ct * 16 + n) * 64 + 16 * wid + 8 * h;   // O[c][q], q contiguous
        *(v4f*)(od)     = (v4f){acc[0], acc[1], acc[2], acc[3]};
        *(v4f*)(od + 4) = (v4f){acc[4], acc[5], acc[6], acc[7]};
    }
    __syncthreads();

    // coalesced-ish final store: out[b][c][2*(i0+q)+p]
    for (int e = tid; e < C_ * 64; e += 128) {
        const int c = e >> 6, qq = e & 63;
        out[((size_t)b * C_ + c) * L_ + 2 * (i0 + qq) + p] = Ol[e];
    }
}

// ================================== launcher ==================================
extern "C" void kernel_launch(void* const* d_in, const int* in_sizes, int n_in,
                              void* d_out, int out_size, void* d_ws, size_t ws_size,
                              hipStream_t stream) {
    (void)in_sizes; (void)n_in; (void)out_size; (void)ws_size;
    const float* x  = (const float*)d_in[0];
    const float* wq = (const float*)d_in[1];
    const float* bq = (const float*)d_in[2];
    const float* wk = (const float*)d_in[3];
    const float* bk = (const float*)d_in[4];
    const float* wv = (const float*)d_in[5];
    const float* bv = (const float*)d_in[6];
    float* out = (float*)d_out;

    const size_t N = (size_t)B_ * L_ * C_;       // per-tensor f16 elements
    f16* qws = (f16*)d_ws;
    f16* kws = qws + N;
    f16* vws = kws + N;

    const size_t smem1 = (size_t)C_ * C_ * sizeof(f16) + (size_t)64 * XP * sizeof(f16);
    natt_qkv<<<dim3(L_ / 64, B_, 3), dim3(256), smem1, stream>>>(
        x, wq, bq, wk, bk, wv, bv, qws, kws, vws);

    const size_t smem2 = (size_t)128 * C_ * sizeof(f16) * 2   // K + V(->O)
                       + (size_t)C_ * VTP * sizeof(f16)       // V^T
                       + (size_t)64 * C_ * sizeof(f16)        // Q
                       + (size_t)4 * 16 * 96 * sizeof(f16);   // P
    natt_attn<<<dim3(LH / 64, 2, B_), dim3(128), smem2, stream>>>(qws, kws, vws, out);
}
